// SNN_19911468384643
// MI455X (gfx1250) — matrix operand order repaired and checked by reading, not compiled, not run on hardware
//
#include <hip/hip_runtime.h>

typedef __attribute__((ext_vector_type(16))) _Float16 v16h;
typedef __attribute__((ext_vector_type(8)))  _Float16 v8h;
typedef __attribute__((ext_vector_type(8)))  float    v8f;

#if defined(__has_builtin)
#if __has_builtin(__builtin_amdgcn_global_load_async_to_lds_b128)
#define HAVE_ASYNC_LDS 1
#endif
#endif

#if defined(HAVE_ASYNC_LDS)
typedef int i32x4 __attribute__((vector_size(16)));
typedef __attribute__((address_space(1))) i32x4* as1_i32x4p;
typedef __attribute__((address_space(3))) i32x4* as3_i32x4p;
#endif

// ---------------- problem constants ----------------
#define NT      15
#define C1      30
#define C1PAD   32           // channel-last spike map padded to 32 (ci 30,31 = 0)
#define C2      250
#define HIN     252
#define HPOOL   128          // padded pooled spike map (126 + 1 border each side)
#define H2      126
#define NPIX    (126*126)    // 15876
#define K2PAD   288          // 9 chunks of 32: k = (kh*3+kw)*32 + ci
#define NKC     9
#define C2PAD   256
#define NOUT1   (NT*C2*NPIX) // 59,535,000 (spk), same for pot
#define NPART   1024

__device__ __forceinline__ void wait_async0() {
#if defined(HAVE_ASYNC_LDS)
#if __has_builtin(__builtin_amdgcn_s_wait_asynccnt)
    __builtin_amdgcn_s_wait_asynccnt(0);
#else
    asm volatile("s_wait_asynccnt 0" ::: "memory");
#endif
#endif
}

#if defined(HAVE_ASYNC_LDS)
// issue 4 async b128 global->LDS transfers covering 64 contiguous bytes
__device__ __forceinline__ void async_copy64(const _Float16* src, _Float16* dst) {
    as1_i32x4p g = (as1_i32x4p)(void*)src;
    as3_i32x4p l = (as3_i32x4p)(void*)dst;
    __builtin_amdgcn_global_load_async_to_lds_b128(g, l, 0,  0);
    __builtin_amdgcn_global_load_async_to_lds_b128(g, l, 16, 0);
    __builtin_amdgcn_global_load_async_to_lds_b128(g, l, 32, 0);
    __builtin_amdgcn_global_load_async_to_lds_b128(g, l, 48, 0);
}
#endif

// =====================================================================
// 0) misc init (global-max scalar must start at 0 every call)
// =====================================================================
__global__ void k_init(unsigned int* gm) {
    if (threadIdx.x == 0 && blockIdx.x == 0) *gm = 0u;
}

// =====================================================================
// 1) repack w2 -> f16 [256][288], K reordered as (kh*3+kw)*32 + ci
// =====================================================================
__global__ void k_prep_w2h(const float* __restrict__ w2, _Float16* __restrict__ w2h) {
    int idx = blockIdx.x * blockDim.x + threadIdx.x;
    if (idx >= C2PAD * K2PAD) return;
    const int c  = idx / K2PAD;
    const int kk = idx % K2PAD;
    const int kc = kk / 32;          // kh*3+kw
    const int ci = kk % 32;
    float v = (c < C2 && ci < C1) ? w2[((size_t)c * C1 + ci) * 9 + kc] : 0.0f;
    w2h[idx] = (_Float16)v;
}

// =====================================================================
// 2) conv1 + fire(15) + maxpool2x2 + pad(1) -> f16 spikes, CHANNEL-LAST
//    spkpool[t][y(128)][x(128)][ci(32)], ci>=30 zero
// =====================================================================
__global__ void k_conv1_fire_pool(const int* __restrict__ inp,
                                  const float* __restrict__ w1,
                                  _Float16* __restrict__ spkpool) {
    const int t = blockIdx.z;
    const int c = blockIdx.y;                 // 0..31 (30,31 are zero-pad channels)
    const int ty = blockIdx.x >> 3, tx = blockIdx.x & 7;
    const int ly = threadIdx.x >> 4, lx = threadIdx.x & 15;
    const int pyo = ty * 16 + ly;             // 0..127 padded pooled coords
    const int pxo = tx * 16 + lx;
    _Float16 outv = (_Float16)0.0f;
    if (c < C1 && pyo >= 1 && pyo <= 126 && pxo >= 1 && pxo <= 126) {
        const int y0 = 2 * (pyo - 1);
        const int x0 = 2 * (pxo - 1);
        float s0 = 0.f, s1 = 0.f, s2 = 0.f, s3 = 0.f;
        for (int ci = 0; ci < 6; ++ci) {
            const int*   ip = inp + ((t * 6 + ci) * HIN * HIN);
            const float* wp = w1  + ((c * 6 + ci) * 25);
            float win[36];
            #pragma unroll
            for (int r = 0; r < 6; ++r) {
                const int iy = y0 - 2 + r;
                #pragma unroll
                for (int cc = 0; cc < 6; ++cc) {
                    const int ix = x0 - 2 + cc;
                    win[r * 6 + cc] = ((unsigned)iy < (unsigned)HIN &&
                                       (unsigned)ix < (unsigned)HIN)
                                          ? (float)ip[iy * HIN + ix] : 0.0f;
                }
            }
            #pragma unroll
            for (int kh = 0; kh < 5; ++kh) {
                #pragma unroll
                for (int kw = 0; kw < 5; ++kw) {
                    const float w = wp[kh * 5 + kw];
                    s0 = fmaf(w, win[kh * 6 + kw],           s0);
                    s1 = fmaf(w, win[kh * 6 + kw + 1],       s1);
                    s2 = fmaf(w, win[(kh + 1) * 6 + kw],     s2);
                    s3 = fmaf(w, win[(kh + 1) * 6 + kw + 1], s3);
                }
            }
        }
        const float mx = fmaxf(fmaxf(s0, s1), fmaxf(s2, s3));
        outv = (mx > 15.0f) ? (_Float16)1.0f : (_Float16)0.0f;
    }
    spkpool[(((size_t)t * HPOOL + pyo) * HPOOL + pxo) * C1PAD + c] = outv;
}

// =====================================================================
// 3) conv2 as implicit GEMM via WMMA f32_16x16x32_f16, fused fire(10)
//    Block: 128 threads = 4 waves. Wave w -> channel tile (blockIdx.x*4+w).
//    K-chunk kc <-> kernel tap (kh,kw); 32 contraction lanes = 32 channels.
//    Double-buffered LDS B staging: fetch tap kc+1 (async-to-LDS when
//    available, else 4x b128 via VGPRs) while computing tap kc.
// =====================================================================
__global__ void __launch_bounds__(128)
k_conv2_wmma(const _Float16* __restrict__ spkpool,
             const _Float16* __restrict__ w2h,
             float* __restrict__ potbuf) {
    __shared__ __align__(64) _Float16 ldsB[2][128 * 32];   // 2 x 8 KB: [pixel][ci]

    const int t     = blockIdx.z;
    const int pbase = blockIdx.y * 128;                  // pixel-group base
    const int wave  = threadIdx.x >> 5;
    const int lane  = threadIdx.x & 31;
    const int cbase = (blockIdx.x * 4 + wave) * 16;      // 0..240

    v8f acc[8];
    #pragma unroll
    for (int i = 0; i < 8; ++i)
        #pragma unroll
        for (int r = 0; r < 8; ++r) acc[i][r] = 0.0f;

    // this thread's pixel for cooperative B staging (clamped; OOB results
    // are never stored, so the loaded values don't matter)
    const int p_off = threadIdx.x;                       // 0..127
    int p = pbase + p_off;
    if (p >= NPIX) p = NPIX - 1;
    const int y = p / H2;
    const int x = p % H2;
    const _Float16* spkt = spkpool + (size_t)t * HPOOL * HPOOL * C1PAD;
    // base of this pixel's 32-channel vector at tap (0,0)
    const _Float16* pixbase = spkt + ((size_t)y * HPOOL + x) * C1PAD;

    // ---- stage tap 0 into buffer 0 ----
    {
        const _Float16* src = pixbase;                   // kh=0,kw=0
        _Float16* dsth = &ldsB[0][p_off * 32];
#if defined(HAVE_ASYNC_LDS)
        async_copy64(src, dsth);
        wait_async0();
#else
        const uint4* s4 = (const uint4*)src;
        uint4* d4 = (uint4*)dsth;
        uint4 r0 = s4[0], r1 = s4[1], r2 = s4[2], r3 = s4[3];
        d4[0] = r0; d4[1] = r1; d4[2] = r2; d4[3] = r3;
#endif
    }
    __syncthreads();

#if !defined(HAVE_ASYNC_LDS)
    uint4 r0, r1, r2, r3;                                // staging temporaries
#endif

    for (int kc = 0; kc < NKC; ++kc) {
        const int cur = kc & 1;
        const int nxt = cur ^ 1;

        // ---- begin fetch of tap kc+1 into the other buffer ----
        if (kc + 1 < NKC) {
            const int kh2 = (kc + 1) / 3, kw2 = (kc + 1) % 3;
            const _Float16* src = pixbase + ((size_t)kh2 * HPOOL + kw2) * C1PAD;
            _Float16* dsth = &ldsB[nxt][p_off * 32];
#if defined(HAVE_ASYNC_LDS)
            async_copy64(src, dsth);
#else
            const uint4* s4 = (const uint4*)src;
            r0 = s4[0]; r1 = s4[1]; r2 = s4[2]; r3 = s4[3];
#endif
        }
        if (kc + 2 < NKC) {                              // prefetch tap kc+2
            const int kh3 = (kc + 2) / 3, kw3 = (kc + 2) % 3;
            __builtin_prefetch(pixbase + ((size_t)kh3 * HPOOL + kw3) * C1PAD, 0, 3);
        }

        // ---- A fragment (16x32, contiguous k order, same as B) ----
        // lane<16: M=lane,    K = {0..15}
        // lane>=16:M=lane-16, K = {16..31}
        const int m  = lane & 15;
        const int kb = (lane >> 4) * 16;
        const _Float16* wrow = w2h + (size_t)(cbase + m) * K2PAD + kc * 32;
        const v8h lo = *(const v8h*)(wrow + kb);
        const v8h hi = *(const v8h*)(wrow + kb + 8);
        v16h a;
        #pragma unroll
        for (int e = 0; e < 8; ++e) { a[e] = lo[e]; a[e + 8] = hi[e]; }

        // ---- B fragments from LDS (lanes 0-15: K0..15, lanes 16-31: K16..31) ----
        const int n  = lane & 15;
        const int kB = (lane >> 4) * 16;
        #pragma unroll
        for (int pt = 0; pt < 8; ++pt) {
            const v16h b = *(const v16h*)(&ldsB[cur][(pt * 16 + n) * 32 + kB]);
            acc[pt] = __builtin_amdgcn_wmma_f32_16x16x32_f16(
                false, a, false, b, (short)0, acc[pt], false, false);
        }

        // ---- commit next buffer, then block-wide handshake ----
        if (kc + 1 < NKC) {
#if defined(HAVE_ASYNC_LDS)
            wait_async0();
#else
            uint4* d4 = (uint4*)(&ldsB[nxt][p_off * 32]);
            d4[0] = r0; d4[1] = r1; d4[2] = r2; d4[3] = r3;
#endif
            __syncthreads();
        }
    }

    // ---- store D with fused fire(10): VGPR r -> M = r (+8 for hi lanes) ----
    const int n  = lane & 15;
    const int mo = (lane >> 4) * 8;
    #pragma unroll
    for (int pt = 0; pt < 8; ++pt) {
        const int pp = pbase + pt * 16 + n;
        if (pp < NPIX) {
            #pragma unroll
            for (int r = 0; r < 8; ++r) {
                const int c = cbase + mo + r;
                if (c < C2) {
                    const float v = acc[pt][r];
                    potbuf[((size_t)t * C2 + c) * NPIX + pp] = (v > 10.0f) ? v : 0.0f;
                }
            }
        }
    }
}

// =====================================================================
// 4) pointwise inhibition, stage 1: per (t,p) max/argmax over C
// =====================================================================
__global__ void k_inhib1(const float* __restrict__ potbuf,
                         float* __restrict__ maxval, int* __restrict__ maxidx) {
    const int idx = blockIdx.x * blockDim.x + threadIdx.x;
    if (idx >= NT * NPIX) return;
    const int t = idx / NPIX, p = idx % NPIX;
    const float* base = potbuf + (size_t)t * C2 * NPIX + p;
    float mv = base[0];
    int   mi = 0;
    for (int c = 1; c < C2; ++c) {
        const float v = base[(size_t)c * NPIX];
        if (v > mv) { mv = v; mi = c; }
    }
    maxval[idx] = mv;
    maxidx[idx] = mi;
}

// 5) stage 2: per p -> winning feature + fired-at-last-step flag
__global__ void k_inhib2(const float* __restrict__ maxval, const int* __restrict__ maxidx,
                         int* __restrict__ winfeat, float* __restrict__ firedlast) {
    const int p = blockIdx.x * blockDim.x + threadIdx.x;
    if (p >= NPIX) return;
    int cnt = 0;
    for (int t = 0; t < NT; ++t) cnt += (maxval[t * NPIX + p] > 0.0f) ? 1 : 0;
    int e = NT - cnt; e = e < 0 ? 0 : (e > NT - 1 ? NT - 1 : e);
    winfeat[p]   = maxidx[e * NPIX + p];
    firedlast[p] = (maxval[(NT - 1) * NPIX + p] > 0.0f) ? 1.0f : 0.0f;
}

// 6) apply inhibition, write final pot + spk to d_out
__global__ void k_inhib_apply(const float* __restrict__ potbuf,
                              const int* __restrict__ winfeat,
                              const float* __restrict__ firedlast,
                              float* __restrict__ out_spk, float* __restrict__ out_pot) {
    const long long idx = (long long)blockIdx.x * blockDim.x + threadIdx.x;
    if (idx >= (long long)NOUT1) return;
    const int p  = (int)(idx % NPIX);
    const int c  = (int)((idx / NPIX) % C2);
    const float coef = (c == winfeat[p]) ? firedlast[p] : 0.0f;
    const float pv   = potbuf[idx] * coef;
    out_pot[idx] = pv;
    out_spk[idx] = (pv > 0.0f) ? 1.0f : 0.0f;
}

// =====================================================================
// 7) k-winners: per (c,p) spike count + value at first-spike time,
//    block-reduce toward the global max of spiking values
// =====================================================================
__global__ void k_win_val(const float* __restrict__ out_pot,
                          float* __restrict__ valbuf, int* __restrict__ nspkbuf,
                          unsigned int* __restrict__ gm) {
    __shared__ float red[256];
    const int idx = blockIdx.x * blockDim.x + threadIdx.x;
    float cand = 0.0f;
    if (idx < C2 * NPIX) {
        const int c = idx / NPIX, p = idx % NPIX;
        int nspk = 0;
        for (int t = 0; t < NT; ++t)
            nspk += (out_pot[((size_t)t * C2 + c) * NPIX + p] > 0.0f) ? 1 : 0;
        int e = NT - nspk; e = e < 0 ? 0 : (e > NT - 1 ? NT - 1 : e);
        const float value = out_pot[((size_t)e * C2 + c) * NPIX + p];
        valbuf[idx]  = value;
        nspkbuf[idx] = nspk;
        cand = (nspk > 0) ? value : 0.0f;   // values are >= 0
    }
    red[threadIdx.x] = cand;
    __syncthreads();
    for (int s = 128; s > 0; s >>= 1) {
        if (threadIdx.x < s) red[threadIdx.x] = fmaxf(red[threadIdx.x], red[threadIdx.x + s]);
        __syncthreads();
    }
    if (threadIdx.x == 0) atomicMax(gm, __float_as_uint(red[0]));
}

// 8) total = nspk * (value + T * globalmax)
__global__ void k_win_total(const float* __restrict__ valbuf, const int* __restrict__ nspkbuf,
                            const unsigned int* __restrict__ gm, float* __restrict__ total) {
    const int idx = blockIdx.x * blockDim.x + threadIdx.x;
    if (idx >= C2 * NPIX) return;
    const float v = __uint_as_float(*gm) * (float)NT;
    total[idx] = (float)nspkbuf[idx] * (valbuf[idx] + v);
}

// 9) two-stage global argmax (first-occurrence ties) + NMS zeroing
__global__ void k_argmax_part(const float* __restrict__ total,
                              float* __restrict__ pval, int* __restrict__ pidx) {
    __shared__ float sv[256];
    __shared__ int   si[256];
    float bv = -1e30f; int bi = -1;
    for (int i = blockIdx.x * blockDim.x + threadIdx.x; i < C2 * NPIX;
         i += gridDim.x * blockDim.x) {
        const float v = total[i];
        if (v > bv) { bv = v; bi = i; }           // i increasing -> first occurrence
    }
    sv[threadIdx.x] = bv; si[threadIdx.x] = bi;
    __syncthreads();
    for (int s = 128; s > 0; s >>= 1) {
        if (threadIdx.x < s) {
            const float ov = sv[threadIdx.x + s]; const int oi = si[threadIdx.x + s];
            if (ov > sv[threadIdx.x] ||
                (ov == sv[threadIdx.x] && oi >= 0 &&
                 (si[threadIdx.x] < 0 || oi < si[threadIdx.x]))) {
                sv[threadIdx.x] = ov; si[threadIdx.x] = oi;
            }
        }
        __syncthreads();
    }
    if (threadIdx.x == 0) { pval[blockIdx.x] = sv[0]; pidx[blockIdx.x] = si[0]; }
}

__global__ void k_argmax_final(const float* __restrict__ pval, const int* __restrict__ pidx,
                               float* __restrict__ out_win, int iter,
                               int* __restrict__ wininfo) {
    __shared__ float sv[256];
    __shared__ int   si[256];
    float bv = -1e30f; int bi = -1;
    for (int i = threadIdx.x; i < NPART; i += 256) {
        const float v = pval[i]; const int ix = pidx[i];
        if (v > bv || (v == bv && ix >= 0 && (bi < 0 || ix < bi))) { bv = v; bi = ix; }
    }
    sv[threadIdx.x] = bv; si[threadIdx.x] = bi;
    __syncthreads();
    for (int s = 128; s > 0; s >>= 1) {
        if (threadIdx.x < s) {
            const float ov = sv[threadIdx.x + s]; const int oi = si[threadIdx.x + s];
            if (ov > sv[threadIdx.x] ||
                (ov == sv[threadIdx.x] && oi >= 0 &&
                 (si[threadIdx.x] < 0 || oi < si[threadIdx.x]))) {
                sv[threadIdx.x] = ov; si[threadIdx.x] = oi;
            }
        }
        __syncthreads();
    }
    if (threadIdx.x == 0) {
        const int wi = si[0] < 0 ? 0 : si[0];
        const float wv = sv[0];
        const int c = wi / NPIX, rem = wi % NPIX, row = rem / H2, col = rem % H2;
        wininfo[0] = c; wininfo[1] = row; wininfo[2] = col;
        float* w3 = out_win + iter * 3;
        if (wv > 0.0f) { w3[0] = (float)c; w3[1] = (float)row; w3[2] = (float)col; }
        else           { w3[0] = -1.0f;    w3[1] = -1.0f;      w3[2] = -1.0f; }
    }
}

__global__ void k_nms_zero(float* __restrict__ total, const int* __restrict__ wininfo) {
    const int idx = blockIdx.x * blockDim.x + threadIdx.x;
    if (idx >= C2 * NPIX) return;
    const int c = idx / NPIX, rem = idx % NPIX, row = rem / H2, col = rem % H2;
    const int wc = wininfo[0], wr = wininfo[1], wl = wininfo[2];
    const int dr = row - wr, dc = col - wl;
    if (c == wc || (dr >= -1 && dr <= 1 && dc >= -1 && dc <= 1)) total[idx] = 0.0f;
}

// =====================================================================
// host-side orchestration
// =====================================================================
static inline size_t align256(size_t x) { return (x + 255) & ~(size_t)255; }

extern "C" void kernel_launch(void* const* d_in, const int* in_sizes, int n_in,
                              void* d_out, int out_size, void* d_ws, size_t ws_size,
                              hipStream_t stream) {
    const int*   inp = (const int*)d_in[0];
    // d_in[1] = max_layer scalar (unused; training path max_layer==2)
    const float* w1  = (const float*)d_in[2];
    const float* w2  = (const float*)d_in[3];
    float* out       = (float*)d_out;
    char*  ws        = (char*)d_ws;

    size_t off = 0;
    _Float16* spkpool = ( _Float16*)(ws + off); off = align256(off + (size_t)NT * HPOOL * HPOOL * C1PAD * 2);
    _Float16* w2h     = ( _Float16*)(ws + off); off = align256(off + (size_t)C2PAD * K2PAD * 2);
    float*    potbuf  = (float*)(ws + off);     off = align256(off + (size_t)NOUT1 * 4);
    float*    maxval  = (float*)(ws + off);     off = align256(off + (size_t)NT * NPIX * 4);
    int*      maxidx  = (int*)(ws + off);       off = align256(off + (size_t)NT * NPIX * 4);
    int*      winfeat = (int*)(ws + off);       off = align256(off + (size_t)NPIX * 4);
    float*    firedl  = (float*)(ws + off);     off = align256(off + (size_t)NPIX * 4);
    float*    valbuf  = (float*)(ws + off);     off = align256(off + (size_t)C2 * NPIX * 4);
    int*      nspkbuf = (int*)(ws + off);       off = align256(off + (size_t)C2 * NPIX * 4);
    float*    total   = (float*)(ws + off);     off = align256(off + (size_t)C2 * NPIX * 4);
    unsigned int* gm  = (unsigned int*)(ws + off); off = align256(off + 256);
    float*    pval    = (float*)(ws + off);     off = align256(off + (size_t)NPART * 4);
    int*      pidx    = (int*)(ws + off);       off = align256(off + (size_t)NPART * 4);
    int*      wininfo = (int*)(ws + off);       off = align256(off + 256);
    (void)ws_size; (void)in_sizes; (void)n_in; (void)out_size;

    float* out_spk = out;
    float* out_pot = out + (size_t)NOUT1;
    float* out_win = out + (size_t)2 * NOUT1;

    k_init<<<1, 64, 0, stream>>>(gm);
    k_prep_w2h<<<(C2PAD * K2PAD + 255) / 256, 256, 0, stream>>>(w2, w2h);
    k_conv1_fire_pool<<<dim3(64, C1PAD, NT), 256, 0, stream>>>(inp, w1, spkpool);
    k_conv2_wmma<<<dim3(4, 125, NT), 128, 0, stream>>>(spkpool, w2h, potbuf);
    k_inhib1<<<(NT * NPIX + 255) / 256, 256, 0, stream>>>(potbuf, maxval, maxidx);
    k_inhib2<<<(NPIX + 255) / 256, 256, 0, stream>>>(maxval, maxidx, winfeat, firedl);
    k_inhib_apply<<<(int)(((long long)NOUT1 + 255) / 256), 256, 0, stream>>>(
        potbuf, winfeat, firedl, out_spk, out_pot);
    k_win_val<<<(C2 * NPIX + 255) / 256, 256, 0, stream>>>(out_pot, valbuf, nspkbuf, gm);
    k_win_total<<<(C2 * NPIX + 255) / 256, 256, 0, stream>>>(valbuf, nspkbuf, gm, total);
    for (int it = 0; it < 8; ++it) {
        k_argmax_part<<<NPART, 256, 0, stream>>>(total, pval, pidx);
        k_argmax_final<<<1, 256, 0, stream>>>(pval, pidx, out_win, it, wininfo);
        k_nms_zero<<<(C2 * NPIX + 255) / 256, 256, 0, stream>>>(total, wininfo);
    }
}
